// ModifiedGCN_70669391888553
// MI455X (gfx1250) — matrix operand order repaired and checked by reading, not compiled, not run on hardware
//
#include <hip/hip_runtime.h>

typedef __attribute__((ext_vector_type(16))) _Float16 v16h;
typedef __attribute__((ext_vector_type(8)))  _Float16 v8h;
typedef __attribute__((ext_vector_type(8)))  float    v8f;

#define N_NODES 50000
#define IN_CH   128
#define HID_CH  256
#define OUT_CH  128
#define N_EDGES 500000
#define N_PAIRS 500000

// ---------------- elementwise / graph kernels ----------------

__global__ void k_fill(float* __restrict__ p, float v, int n) {
  int i = blockIdx.x * blockDim.x + threadIdx.x;
  if (i < n) p[i] = v;
}

__global__ void k_deg_accum(const int* __restrict__ dst, float* __restrict__ deg, int e) {
  int i = blockIdx.x * blockDim.x + threadIdx.x;
  if (i < e) atomicAdd(&deg[dst[i]], 1.0f);
}

__global__ void k_rsqrt(float* __restrict__ p, int n) {
  int i = blockIdx.x * blockDim.x + threadIdx.x;
  if (i < n) p[i] = rsqrtf(p[i]);
}

// 4 edges per 256-thread block; 64 lanes per edge, float4 per lane (256 ch).
__global__ void k_scatter(const int* __restrict__ src, const int* __restrict__ dst,
                          const float* __restrict__ dinv, const float* __restrict__ t,
                          float* __restrict__ agg) {
  int e = blockIdx.x * 4 + (threadIdx.x >> 6);
  if (e >= N_EDGES) return;
  int s = src[e], d = dst[e];
  float norm = dinv[s] * dinv[d];
  int f = (threadIdx.x & 63) * 4;
  const float4 v = *(const float4*)(t + (size_t)s * HID_CH + f);
  float* o = agg + (size_t)d * HID_CH + f;
  atomicAdd(o + 0, v.x * norm);
  atomicAdd(o + 1, v.y * norm);
  atomicAdd(o + 2, v.z * norm);
  atomicAdd(o + 3, v.w * norm);
}

// h[idx] = relu(agg[idx] + t[idx]*dinv[i]^2 + b[f])
__global__ void k_self_bias_relu(const float* __restrict__ t, const float* __restrict__ dinv,
                                 const float* __restrict__ b, float* __restrict__ h) {
  int idx = blockIdx.x * blockDim.x + threadIdx.x;
  if (idx >= N_NODES * HID_CH) return;
  int i = idx >> 8;          // HID_CH == 256
  int f = idx & 255;
  float di = dinv[i];
  float v = h[idx] + t[idx] * di * di + b[f];
  h[idx] = fmaxf(v, 0.0f);
}

// ---------------- WMMA helpers ----------------

__device__ __forceinline__ v8h pack8(float4 a, float4 b) {
  v8h r;
  r[0] = (_Float16)a.x; r[1] = (_Float16)a.y; r[2] = (_Float16)a.z; r[3] = (_Float16)a.w;
  r[4] = (_Float16)b.x; r[5] = (_Float16)b.y; r[6] = (_Float16)b.z; r[7] = (_Float16)b.w;
  return r;
}

// Fragment-major LDS tiles (ISA 7.12.2 layouts), 64x64 block tile, K-step 32.
// A frag record: m-subtile s (0..3), lane L -> row m = s*16 + (L&15),
//   halves j: K = 16*(j>>3) + 8*(L>>4) + (j&7)
// B frag record: n-subtile s (0..3), lane L -> col n = s*16 + (L&15),
//   halves j: K = 16*(j>>3) + 8*(L>>4) + (j&7)

#define BM 64
#define BN 64
#define BK 32

// ---------------- GEMM: C[M,N] = A[M,K] @ B[K,N] (fp32 io, f16 WMMA, f32 acc) ----------------

__global__ __launch_bounds__(256)
void k_gemm(const float* __restrict__ A, const float* __restrict__ B,
            float* __restrict__ C, int M, int K, int N) {
  __shared__ __align__(32) _Float16 Af[4 * 32 * 16];
  __shared__ __align__(32) _Float16 Bf[4 * 32 * 16];

  const int tid  = threadIdx.x;
  const int lane = tid & 31;
  const int wave = tid >> 5;
  const int wmi  = wave & 3;   // m-subtile
  const int wni  = wave >> 2;  // n-subtile pair: {2*wni, 2*wni+1}
  const int block_m = blockIdx.y * BM;
  const int block_n = blockIdx.x * BN;
  const int l16 = lane & 15, lhi = lane >> 4;

  // A staging: thread -> (row ar, k-run ac..ac+7); maps to one contiguous 8-half frag run
  const int ar = tid >> 2;
  const int ac = (tid & 3) * 8;
  const int a_gm  = block_m + ar;
  const bool a_ok = a_gm < M;
  const int a_gmc = a_ok ? a_gm : 0;
  _Float16* a_dst = &Af[(((ar >> 4) * 32) + (ar & 15) + ((ac >> 3) & 1) * 16) * 16 + (ac >> 4) * 8];
  const float* a_src = A + (size_t)a_gmc * K + ac;

  // B staging: thread -> (col bn, k-run bkg..bkg+7); contiguous 8-half frag run
  const int bn  = tid & 63;
  const int bkg = (tid >> 6) * 8;
  _Float16* b_dst = &Bf[(((bn >> 4) * 32) + (bn & 15) + ((bkg >> 3) & 1) * 16) * 16 + (bkg >> 4) * 8];
  const float* b_src = B + (size_t)bkg * N + block_n + bn;

  v8f acc0 = {}, acc1 = {};

  for (int k0 = 0; k0 < K; k0 += BK) {
    float4 av0 = *(const float4*)(a_src + k0);
    float4 av1 = *(const float4*)(a_src + k0 + 4);
    if (!a_ok) { av0 = make_float4(0.f, 0.f, 0.f, 0.f); av1 = av0; }
    float bv[8];
#pragma unroll
    for (int j = 0; j < 8; ++j) bv[j] = b_src[(size_t)(k0 + j) * N];

    *(v8h*)a_dst = pack8(av0, av1);
    v8h bp;
#pragma unroll
    for (int j = 0; j < 8; ++j) bp[j] = (_Float16)bv[j];
    *(v8h*)b_dst = bp;
    __syncthreads();

    v16h a  = *(const v16h*)&Af[(wmi * 32 + lane) * 16];
    v16h b0 = *(const v16h*)&Bf[((2 * wni + 0) * 32 + lane) * 16];
    v16h b1 = *(const v16h*)&Bf[((2 * wni + 1) * 32 + lane) * 16];
    acc0 = __builtin_amdgcn_wmma_f32_16x16x32_f16(false, a, false, b0, (short)0, acc0, false, false);
    acc1 = __builtin_amdgcn_wmma_f32_16x16x32_f16(false, a, false, b1, (short)0, acc1, false, false);
    __syncthreads();
  }

  // D: VGPR r -> M = wmi*16 + r + 8*lhi, N = subtile*16 + l16
#pragma unroll
  for (int r = 0; r < 8; ++r) {
    int gm = block_m + wmi * 16 + r + 8 * lhi;
    if (gm < M) {
      C[(size_t)gm * N + block_n + wni * 32 + l16]      = acc0[r];
      C[(size_t)gm * N + block_n + wni * 32 + 16 + l16] = acc1[r];
    }
  }
}

// ---------------- fused pair head: out[p] = concat(h[pa],h[pb]) @ Wfc + bfc ----------------
// M = N_PAIRS, K = 512 (gathered rows), N = 128.

__global__ __launch_bounds__(256)
void k_pair_gemm(const float* __restrict__ H, const int* __restrict__ pairs,
                 const float* __restrict__ Wfc, const float* __restrict__ bfc,
                 float* __restrict__ out, int M) {
  __shared__ __align__(32) _Float16 Af[4 * 32 * 16];
  __shared__ __align__(32) _Float16 Bf[4 * 32 * 16];
  const int K = 2 * HID_CH, N = OUT_CH;

  const int tid  = threadIdx.x;
  const int lane = tid & 31;
  const int wave = tid >> 5;
  const int wmi  = wave & 3;
  const int wni  = wave >> 2;
  const int block_m = blockIdx.y * BM;
  const int block_n = blockIdx.x * BN;
  const int l16 = lane & 15, lhi = lane >> 4;

  const int ar = tid >> 2;
  const int ac = (tid & 3) * 8;
  const int a_gm  = block_m + ar;
  const bool a_ok = a_gm < M;
  const int a_gmc = a_ok ? a_gm : 0;
  _Float16* a_dst = &Af[(((ar >> 4) * 32) + (ar & 15) + ((ac >> 3) & 1) * 16) * 16 + (ac >> 4) * 8];

  const int bn  = tid & 63;
  const int bkg = (tid >> 6) * 8;
  _Float16* b_dst = &Bf[(((bn >> 4) * 32) + (bn & 15) + ((bkg >> 3) & 1) * 16) * 16 + (bkg >> 4) * 8];
  const float* b_src = Wfc + (size_t)bkg * N + block_n + bn;

  v8f acc0 = {}, acc1 = {};

  for (int k0 = 0; k0 < K; k0 += BK) {
    // gather: virtual A row = concat(H[pairs[i][0]], H[pairs[i][1]]); 8-run never crosses split
    int kk = k0 + ac;
    int node = pairs[a_gmc * 2 + (kk >> 8)];
    const float* srcp = H + (size_t)node * HID_CH + (kk & 255);
    float4 av0 = *(const float4*)(srcp);
    float4 av1 = *(const float4*)(srcp + 4);
    if (!a_ok) { av0 = make_float4(0.f, 0.f, 0.f, 0.f); av1 = av0; }
    float bv[8];
#pragma unroll
    for (int j = 0; j < 8; ++j) bv[j] = b_src[(size_t)(k0 + j) * N];

    *(v8h*)a_dst = pack8(av0, av1);
    v8h bp;
#pragma unroll
    for (int j = 0; j < 8; ++j) bp[j] = (_Float16)bv[j];
    *(v8h*)b_dst = bp;
    __syncthreads();

    v16h a  = *(const v16h*)&Af[(wmi * 32 + lane) * 16];
    v16h b0 = *(const v16h*)&Bf[((2 * wni + 0) * 32 + lane) * 16];
    v16h b1 = *(const v16h*)&Bf[((2 * wni + 1) * 32 + lane) * 16];
    acc0 = __builtin_amdgcn_wmma_f32_16x16x32_f16(false, a, false, b0, (short)0, acc0, false, false);
    acc1 = __builtin_amdgcn_wmma_f32_16x16x32_f16(false, a, false, b1, (short)0, acc1, false, false);
    __syncthreads();
  }

#pragma unroll
  for (int r = 0; r < 8; ++r) {
    int gm = block_m + wmi * 16 + r + 8 * lhi;
    if (gm < M) {
      int n0 = block_n + wni * 32 + l16;
      int n1 = n0 + 16;
      out[(size_t)gm * N + n0] = acc0[r] + bfc[n0];
      out[(size_t)gm * N + n1] = acc1[r] + bfc[n1];
    }
  }
}

// ---------------- launcher ----------------

extern "C" void kernel_launch(void* const* d_in, const int* in_sizes, int n_in,
                              void* d_out, int out_size, void* d_ws, size_t ws_size,
                              hipStream_t stream) {
  const float* x     = (const float*)d_in[0];
  const int*   ei    = (const int*)d_in[1];
  const int*   src   = ei;
  const int*   dst   = ei + N_EDGES;
  const int*   pairs = (const int*)d_in[2];
  const float* W1  = (const float*)d_in[3];
  const float* b1  = (const float*)d_in[4];
  const float* W2  = (const float*)d_in[5];
  const float* b2  = (const float*)d_in[6];
  const float* Wfc = (const float*)d_in[7];
  const float* bfc = (const float*)d_in[8];
  float* out = (float*)d_out;

  float* dinv = (float*)d_ws;
  float* bufA = dinv + 50176;                       // 256-aligned
  float* bufB = bufA + (size_t)N_NODES * HID_CH;

  const int NF = N_NODES * HID_CH;

  // degree (self-loop -> init 1.0) -> dinv
  k_fill<<<(N_NODES + 255) / 256, 256, 0, stream>>>(dinv, 1.0f, N_NODES);
  k_deg_accum<<<(N_EDGES + 255) / 256, 256, 0, stream>>>(dst, dinv, N_EDGES);
  k_rsqrt<<<(N_NODES + 255) / 256, 256, 0, stream>>>(dinv, N_NODES);

  dim3 gHid(HID_CH / BN, (N_NODES + BM - 1) / BM);

  // layer 1
  k_gemm<<<gHid, 256, 0, stream>>>(x, W1, bufA, N_NODES, IN_CH, HID_CH);
  k_fill<<<(NF + 255) / 256, 256, 0, stream>>>(bufB, 0.0f, NF);
  k_scatter<<<(N_EDGES + 3) / 4, 256, 0, stream>>>(src, dst, dinv, bufA, bufB);
  k_self_bias_relu<<<(NF + 255) / 256, 256, 0, stream>>>(bufA, dinv, b1, bufB);

  // layer 2
  k_gemm<<<gHid, 256, 0, stream>>>(bufB, W2, bufA, N_NODES, HID_CH, HID_CH);
  k_fill<<<(NF + 255) / 256, 256, 0, stream>>>(bufB, 0.0f, NF);
  k_scatter<<<(N_EDGES + 3) / 4, 256, 0, stream>>>(src, dst, dinv, bufA, bufB);
  k_self_bias_relu<<<(NF + 255) / 256, 256, 0, stream>>>(bufA, dinv, b2, bufB);

  // fused pair head (gather-GEMM)
  dim3 gOut(OUT_CH / BN, (N_PAIRS + BM - 1) / BM);
  k_pair_gemm<<<gOut, 256, 0, stream>>>(bufB, pairs, Wfc, bfc, out, N_PAIRS);
}
